// StackEncoder_26147760898496
// MI455X (gfx1250) — matrix operand order repaired and checked by reading, not compiled, not run on hardware
//
#include <hip/hip_runtime.h>
#include <stdint.h>
#include <stddef.h>

#define B    128
#define L    128
#define H    512
#define T    255      // 2*L-1
#define SMAX 130      // L+2
#define KDIM 1024     // 2*H (h_l || h_r)
#define N5   2560     // 5*H
#define GRIDN 32      // persistent grid: H/16 k-tiles
#define BLK   256     // 8 waves = 8 M-tiles of B

typedef __attribute__((ext_vector_type(16))) __bf16 v16bf;
typedef __attribute__((ext_vector_type(8)))  __bf16 v8bf;
typedef __attribute__((ext_vector_type(8)))  float  v8f;

__device__ __forceinline__ float sigm(float x)  { return 1.0f / (1.0f + __expf(-x)); }
__device__ __forceinline__ float tanhr(float x) { return 1.0f - 2.0f / (__expf(2.0f * x) + 1.0f); }
__device__ __forceinline__ __bf16 bhi(float x)  { return (__bf16)x; }
__device__ __forceinline__ __bf16 blo(float x)  { __bf16 h = (__bf16)x; return (__bf16)(x - (float)h); }

// ---------------------------------------------------------------------------
// One-time: transpose + hi/lo bf16 split of [W_ul ; W_ur] -> Wt[N5][KDIM]
// ---------------------------------------------------------------------------
__global__ void conv_weights(const float* __restrict__ Wul, const float* __restrict__ Wur,
                             __bf16* __restrict__ Wt_hi, __bf16* __restrict__ Wt_lo) {
    int idx = blockIdx.x * blockDim.x + threadIdx.x;   // n * KDIM + k
    int n = idx / KDIM, k = idx % KDIM;
    float w = (k < H) ? Wul[(size_t)k * N5 + n] : Wur[(size_t)(k - H) * N5 + n];
    Wt_hi[idx] = bhi(w);
    Wt_lo[idx] = blo(w);
}

// ---------------------------------------------------------------------------
// One-time: per-batch-row sequential scan of transitions -> ptr/bptr per step
// ---------------------------------------------------------------------------
__global__ void scan_kernel(const int* __restrict__ trans,
                            int* __restrict__ ptrs, int* __restrict__ bptrs) {
    int b = threadIdx.x;                 // 128 threads
    int ptr = 2, bp = 0;
    for (int t = 0; t < T; ++t) {
        ptrs[t * B + b] = ptr;
        bptrs[t * B + b] = bp;
        if (trans[b * T + t] == 1) { ptr += 1; bp += 1; } else { ptr -= 1; }
    }
    ptrs[T * B + b] = ptr;
    bptrs[T * B + b] = bp;
}

// ---------------------------------------------------------------------------
// One-time: init stack rows 0/1 with token 0, build dense step-0 operands,
// and reset the grid-barrier counter (ws is poisoned; must re-zero per call).
// ---------------------------------------------------------------------------
__global__ void init_kernel(const float* __restrict__ seq,
                            float* __restrict__ sh, float* __restrict__ sc,
                            __bf16* __restrict__ A_hi, __bf16* __restrict__ A_lo,
                            float* __restrict__ clb, float* __restrict__ crb,
                            int* __restrict__ bar_cnt) {
    int idx = blockIdx.x * blockDim.x + threadIdx.x;   // B*H
    if (idx == 0) *bar_cnt = 0;
    int b = idx / H, k = idx % H;
    float th = seq[((size_t)b * L + 0) * (2 * H) + k];
    float tc = seq[((size_t)b * L + 0) * (2 * H) + H + k];
    sh[((size_t)b * SMAX + 0) * H + k] = th;
    sh[((size_t)b * SMAX + 1) * H + k] = th;
    sc[((size_t)b * SMAX + 0) * H + k] = tc;
    sc[((size_t)b * SMAX + 1) * H + k] = tc;
    __bf16 h = bhi(th), l = blo(th);
    A_hi[(size_t)b * KDIM + k]     = h;  A_hi[(size_t)b * KDIM + H + k] = h;
    A_lo[(size_t)b * KDIM + k]     = l;  A_lo[(size_t)b * KDIM + H + k] = l;
    clb[idx] = tc;  crb[idx] = tc;
}

// ---------------------------------------------------------------------------
// Device-wide sense-free barrier: monotone generation counter in L2.
// Release fence flushes this WGP's stores, acquire fence invalidates L0 so
// the next phase sees other WGPs' data.
// ---------------------------------------------------------------------------
__device__ __forceinline__ void grid_sync(int* cnt, int gen) {
    __builtin_amdgcn_fence(__ATOMIC_RELEASE, "agent");
    __syncthreads();
    if (threadIdx.x == 0) {
        __hip_atomic_fetch_add(cnt, 1, __ATOMIC_ACQ_REL, __HIP_MEMORY_SCOPE_AGENT);
        while (__hip_atomic_load(cnt, __ATOMIC_ACQUIRE, __HIP_MEMORY_SCOPE_AGENT)
               < gen * GRIDN) {
            __builtin_amdgcn_s_sleep(1);
        }
    }
    __syncthreads();
    __builtin_amdgcn_fence(__ATOMIC_ACQUIRE, "agent");
}

// ---------------------------------------------------------------------------
// Persistent fused kernel: all 255 shift-reduce steps in one launch.
// Per step:
//   Phase A: (128x1024)@(1024x2560) GEMM via hi/lo-split bf16 WMMA; each wave
//            owns one (M-tile, k-tile) and computes all 5 gate tiles, then the
//            fused Tree-LSTM cell -> pending h_j/c_j.
//   Phase B: commit shift/reduce into the stack, gather next step's dense
//            operands (incl. bf16 hi/lo split). Thread (b,k) only touches
//            column k of batch b -> race-free.
// ---------------------------------------------------------------------------
__global__ void __launch_bounds__(BLK) fused_kernel(
        const __bf16* __restrict__ Wt_hi, const __bf16* __restrict__ Wt_lo,
        __bf16* __restrict__ A_hi,  __bf16* __restrict__ A_lo,
        float* __restrict__ clb, float* __restrict__ crb,
        const float* __restrict__ bias,
        float* __restrict__ pend_h, float* __restrict__ pend_c,
        const int* __restrict__ trans,
        const int* __restrict__ ptrs, const int* __restrict__ bptrs,
        const float* __restrict__ seq,
        float* __restrict__ sh, float* __restrict__ sc,
        float* __restrict__ out,
        int* bar_cnt) {
    const int lane   = threadIdx.x & 31;
    const int mt     = threadIdx.x >> 5;   // M tile 0..7
    const int kt     = blockIdx.x;         // hidden-column tile 0..31
    const int lane15 = lane & 15;
    const int half   = lane >> 4;

    // Loop-invariant fragment addressing.
    // A (contiguous k order, matches B): lanes<16 K={0-15}, lanes>=16 K={16-31}
    const int rowA = mt * 16 + lane15;
    const __bf16* aHbase = A_hi + (size_t)rowA * KDIM + half * 16;
    const __bf16* aLbase = A_lo + (size_t)rowA * KDIM + half * 16;
    const int ncol = kt * 16 + lane15;     // B-fragment / output column
    const float bi  = bias[0 * H + ncol];
    const float bo  = bias[1 * H + ncol];
    const float bfl = bias[2 * H + ncol];
    const float bfr = bias[3 * H + ncol];
    const float bu  = bias[4 * H + ncol];

    int gen = 0;
    for (int t = 0; t < T; ++t) {
        // ---------------- Phase A: GEMM + LSTM cell ----------------
        v8f acc[5] = {};
#pragma unroll 1
        for (int ks = 0; ks < KDIM / 32; ++ks) {
            const int k0 = ks * 32;
            v8bf ah0 = *(const v8bf*)(aHbase + k0);
            v8bf ah1 = *(const v8bf*)(aHbase + k0 + 8);
            v8bf al0 = *(const v8bf*)(aLbase + k0);
            v8bf al1 = *(const v8bf*)(aLbase + k0 + 8);
            v16bf a_hi = __builtin_shufflevector(ah0, ah1, 0,1,2,3,4,5,6,7,8,9,10,11,12,13,14,15);
            v16bf a_lo = __builtin_shufflevector(al0, al1, 0,1,2,3,4,5,6,7,8,9,10,11,12,13,14,15);
#pragma unroll
            for (int g = 0; g < 5; ++g) {
                const size_t boff = ((size_t)(g * H + ncol)) * KDIM + k0 + half * 16;
                v16bf b_hi = *(const v16bf*)(Wt_hi + boff);
                v16bf b_lo = *(const v16bf*)(Wt_lo + boff);
                acc[g] = __builtin_amdgcn_wmma_f32_16x16x32_bf16(false, a_hi, false, b_hi,
                                                                 (short)0, acc[g], false, false);
                acc[g] = __builtin_amdgcn_wmma_f32_16x16x32_bf16(false, a_hi, false, b_lo,
                                                                 (short)0, acc[g], false, false);
                acc[g] = __builtin_amdgcn_wmma_f32_16x16x32_bf16(false, a_lo, false, b_hi,
                                                                 (short)0, acc[g], false, false);
            }
        }

        // C/D layout: element j = row (j + 8*half), column = lane15.
#pragma unroll
        for (int j = 0; j < 8; ++j) {
            const int bb = mt * 16 + half * 8 + j;
            const size_t o = (size_t)bb * H + ncol;
            float gi  = acc[0][j] + bi;
            float go  = acc[1][j] + bo;
            float gfl = acc[2][j] + bfl;
            float gfr = acc[3][j] + bfr;
            float gu  = acc[4][j] + bu;
            float cj = sigm(gi) * tanhr(gu) + sigm(gfl) * clb[o] + sigm(gfr) * crb[o];
            float hj = sigm(go) * tanhr(cj);
            pend_h[o] = hj;
            pend_c[o] = cj;
        }

        grid_sync(bar_cnt, ++gen);

        // ---------------- Phase B: commit + gather ----------------
        const int last = (t == T - 1);
#pragma unroll 1
        for (int i = 0; i < (B * H) / (GRIDN * BLK); ++i) {
            const int idx = (blockIdx.x * BLK + (int)threadIdx.x) + i * (GRIDN * BLK);
            const int b = idx >> 9;            // / H
            const int k = idx & (H - 1);       // % H
            const int shift = (trans[b * T + t] == 1);
            const int ptr = ptrs[t * B + b];
            const int bp  = bptrs[t * B + b];
            const size_t o = (size_t)b * H + k;

            float nh, nc;
            if (shift) {
                nh = seq[((size_t)b * L + bp) * (2 * H) + k];
                nc = seq[((size_t)b * L + bp) * (2 * H) + H + k];
            } else {
                nh = pend_h[o];
                nc = pend_c[o];
            }
            const int pos = shift ? ptr : (ptr - 2);
            sh[((size_t)b * SMAX + pos) * H + k] = nh;
            sc[((size_t)b * SMAX + pos) * H + k] = nc;

            if (last) {        // final result == value just pushed (both cases)
                out[o] = nh;
            } else {
                const int p1 = ptrs[(t + 1) * B + b];
                float hl = sh[((size_t)b * SMAX + p1 - 2) * H + k];
                float hr = sh[((size_t)b * SMAX + p1 - 1) * H + k];
                clb[o] = sc[((size_t)b * SMAX + p1 - 2) * H + k];
                crb[o] = sc[((size_t)b * SMAX + p1 - 1) * H + k];
                A_hi[(size_t)b * KDIM + k]     = bhi(hl);
                A_lo[(size_t)b * KDIM + k]     = blo(hl);
                A_hi[(size_t)b * KDIM + H + k] = bhi(hr);
                A_lo[(size_t)b * KDIM + H + k] = blo(hr);
            }
        }

        grid_sync(bar_cnt, ++gen);
    }
}

// ---------------------------------------------------------------------------
extern "C" void kernel_launch(void* const* d_in, const int* in_sizes, int n_in,
                              void* d_out, int out_size, void* d_ws, size_t ws_size,
                              hipStream_t stream) {
    (void)in_sizes; (void)n_in; (void)out_size; (void)ws_size;
    const float* seq   = (const float*)d_in[0];   // (B, L, 2H) f32
    const int*   trans = (const int*)  d_in[1];   // (B, T)     i32
    const float* Wul   = (const float*)d_in[2];   // (H, 5H)    f32
    const float* bias  = (const float*)d_in[3];   // (5H,)      f32
    const float* Wur   = (const float*)d_in[4];   // (H, 5H)    f32
    float* out = (float*)d_out;                   // (B, H)     f32

    char* ws = (char*)d_ws;
    size_t off = 0;
    auto carve = [&](size_t bytes) -> char* {
        char* p = ws + off;
        off += (bytes + 255) & ~(size_t)255;
        return p;
    };
    __bf16* Wt_hi  = (__bf16*)carve((size_t)N5 * KDIM * sizeof(__bf16));
    __bf16* Wt_lo  = (__bf16*)carve((size_t)N5 * KDIM * sizeof(__bf16));
    float*  sh     = (float*) carve((size_t)B * SMAX * H * sizeof(float));
    float*  sc     = (float*) carve((size_t)B * SMAX * H * sizeof(float));
    __bf16* A_hi   = (__bf16*)carve((size_t)B * KDIM * sizeof(__bf16));
    __bf16* A_lo   = (__bf16*)carve((size_t)B * KDIM * sizeof(__bf16));
    float*  clb    = (float*) carve((size_t)B * H * sizeof(float));
    float*  crb    = (float*) carve((size_t)B * H * sizeof(float));
    float*  pend_h = (float*) carve((size_t)B * H * sizeof(float));
    float*  pend_c = (float*) carve((size_t)B * H * sizeof(float));
    int*    ptrs   = (int*)   carve((size_t)(T + 1) * B * sizeof(int));
    int*    bptrs  = (int*)   carve((size_t)(T + 1) * B * sizeof(int));
    int*    barcnt = (int*)   carve(256);

    conv_weights<<<(N5 * KDIM) / 256, 256, 0, stream>>>(Wul, Wur, Wt_hi, Wt_lo);
    scan_kernel<<<1, B, 0, stream>>>(trans, ptrs, bptrs);
    init_kernel<<<(B * H) / 256, 256, 0, stream>>>(seq, sh, sc, A_hi, A_lo, clb, crb, barcnt);

    fused_kernel<<<dim3(GRIDN), BLK, 0, stream>>>(Wt_hi, Wt_lo, A_hi, A_lo, clb, crb,
                                                  bias, pend_h, pend_c,
                                                  trans, ptrs, bptrs, seq,
                                                  sh, sc, out, barcnt);
}
